// Head_73435350827604
// MI455X (gfx1250) — matrix operand order repaired and checked by reading, not compiled, not run on hardware
//
#include <hip/hip_runtime.h>
#include <math.h>

// ---------------------------------------------------------------------------
// Problem constants (reference): B=4, N_RAW=8192, N1=1024, N2=256,
// N_P=64, N_I=512, EMB=512, EPS=1e-5
// ---------------------------------------------------------------------------
#define BB 4
#define N0 8192
#define NN1 1024
#define NN2 256
#define NP 64
#define EMB 512
#define EPS 1e-5f

typedef __attribute__((ext_vector_type(16))) _Float16 v16h;
typedef __attribute__((ext_vector_type(8)))  _Float16 v8h;
typedef __attribute__((ext_vector_type(8)))  float    v8f;

// ===========================================================================
// 3-NN (brute force) + inverse-distance weights.  xyz layout: (B, 3, N).
// ===========================================================================
__global__ __launch_bounds__(256) void knn3_kernel(
    const float* __restrict__ xyzd, const float* __restrict__ xyzs,
    int Nd, int Ns, int* __restrict__ idx, float* __restrict__ wout)
{
    __shared__ float sx[1024], sy[1024], sz[1024];
    const int b = blockIdx.z;
    const int n = blockIdx.x * 256 + threadIdx.x;
    const float* S = xyzs + (size_t)b * 3 * Ns;
    for (int s = threadIdx.x; s < Ns; s += 256) {
        sx[s] = S[s]; sy[s] = S[Ns + s]; sz[s] = S[2 * Ns + s];
    }
    __syncthreads();
    if (n >= Nd) return;
    const float* D = xyzd + (size_t)b * 3 * Nd;
    const float px = D[n], py = D[Nd + n], pz = D[2 * Nd + n];
    float d0 = 3e38f, d1 = 3e38f, d2 = 3e38f;
    int   i0 = 0,     i1 = 0,     i2 = 0;
    for (int s = 0; s < Ns; ++s) {
        const float dx = px - sx[s], dy = py - sy[s], dz = pz - sz[s];
        const float d = dx * dx + dy * dy + dz * dz;
        if (d < d0)      { d2 = d1; i2 = i1; d1 = d0; i1 = i0; d0 = d; i0 = s; }
        else if (d < d1) { d2 = d1; i2 = i1; d1 = d;  i1 = s; }
        else if (d < d2) { d2 = d;  i2 = s; }
    }
    const float w0 = 1.f / (d0 + 1e-8f);
    const float w1 = 1.f / (d1 + 1e-8f);
    const float w2 = 1.f / (d2 + 1e-8f);
    const float ws = w0 + w1 + w2;
    const size_t o = ((size_t)b * Nd + n) * 3;
    idx[o] = i0; idx[o + 1] = i1; idx[o + 2] = i2;
    wout[o] = w0 / ws; wout[o + 1] = w1 / ws; wout[o + 2] = w2 / ws;
}

// convert multi_feature[:, :64, :] (fp32, (B,576,512)) -> f16 (B,64,512)
__global__ __launch_bounds__(256) void convert_palign_kernel(
    const float* __restrict__ multi, _Float16* __restrict__ P16)
{
    const int i = blockIdx.x * 256 + threadIdx.x;
    if (i >= BB * NP * EMB) return;
    const int b = i / (NP * EMB);
    P16[i] = (_Float16)multi[(size_t)b * 576 * EMB + (i % (NP * EMB))];
}

// ===========================================================================
// interp[b, coff+c, n] = sum_k w[b,n,k] * src(b, idx[b,n,k], c)   (f16 src)
// src element (b, s, c) addressed as src[b*sb + c*sc + s*ss].
// ===========================================================================
__global__ __launch_bounds__(256) void gather_interp_kernel(
    _Float16* __restrict__ xdst, const _Float16* __restrict__ src,
    const int* __restrict__ idx, const float* __restrict__ w,
    int C, int Nd, int Cx, int coff, long sb, long sc, long ss)
{
    const long i = (long)blockIdx.x * 256 + threadIdx.x;
    const long total = (long)BB * C * Nd;
    if (i >= total) return;
    const int  n = (int)(i % Nd);
    const long r = i / Nd;
    const int  c = (int)(r % C);
    const int  b = (int)(r / C);
    const int*   ip = idx + ((long)b * Nd + n) * 3;
    const float* wp = w   + ((long)b * Nd + n) * 3;
    const _Float16* sp = src + (long)b * sb + (long)c * sc;
    const float v = wp[0] * (float)sp[(long)ip[0] * ss] +
                    wp[1] * (float)sp[(long)ip[1] * ss] +
                    wp[2] * (float)sp[(long)ip[2] * ss];
    xdst[((long)b * Cx + coff + c) * Nd + n] = (_Float16)v;
}

// copy skip-connection channels fp32 -> f16: x[b, coff+c, n] = src[b, c, n]
__global__ __launch_bounds__(256) void copy_channels_kernel(
    _Float16* __restrict__ x, const float* __restrict__ src,
    int Csrc, int N, int Cx, int coff)
{
    const long i = (long)blockIdx.x * 256 + threadIdx.x;
    const long total = (long)BB * Csrc * N;
    if (i >= total) return;
    const int  n = (int)(i % N);
    const long r = i / N;
    const int  c = (int)(r % Csrc);
    const int  b = (int)(r / Csrc);
    x[((long)b * Cx + coff + c) * N + n] = (_Float16)src[i];
}

// ===========================================================================
// WMMA f16 GEMM:  Z[b,o,n] = sum_c W[o,c] * X[b,c,n] + bias[o]
// W fp32 (O x C), X f16 (B, C, N), Z fp32 (B, O, N).
// 256 threads = 8 wave32s (4Mx2N); block tile 64x64; wave = 16x32 output via
// two v_wmma_f32_16x16x32_f16 per 32-wide K step.
// Double-buffered LDS, register-staged pipeline, ONE barrier per K tile:
//   load_regs(t+1) -> compute(buf) -> stage_store(buf^1) -> barrier
// O, N multiples of 64; arbitrary C (tail only predicates the A regs / B row).
// ===========================================================================
#define LSTR 40   // LDS row stride in halves (80 B: rows 16B-aligned)

__device__ __forceinline__ void gemm_load_regs(
    const float* __restrict__ Wr, const _Float16* __restrict__ Xb,
    int C, int N, int n0, int k0, int akg, int bkr, int bng,
    float (&af)[8], v8h& bh)
{
    if (k0 + 32 <= C) {                          // uniform bulk path
        #pragma unroll
        for (int j = 0; j < 8; ++j) af[j] = Wr[k0 + akg + j];
        bh = *(const v8h*)(Xb + (size_t)(k0 + bkr) * N + n0 + bng);
        // speculative prefetch of the tile after next (global_prefetch_b8)
        if (k0 + 64 <= C)
            __builtin_prefetch(Xb + (size_t)(k0 + 32 + bkr) * N + n0 + bng, 0, 1);
    } else {                                     // K tail
        #pragma unroll
        for (int j = 0; j < 8; ++j) {
            const int k = k0 + akg + j;
            af[j] = (k < C) ? Wr[k] : 0.f;
        }
        const int kb = k0 + bkr;
        if (kb < C) {
            bh = *(const v8h*)(Xb + (size_t)kb * N + n0 + bng);
        } else {
            #pragma unroll
            for (int j = 0; j < 8; ++j) bh[j] = (_Float16)0.f;
        }
    }
}

__device__ __forceinline__ void gemm_stage_store(
    _Float16* __restrict__ As, _Float16* __restrict__ Bs,
    int ar, int akg, int bkr, int bng, const float (&af)[8], const v8h& bh)
{
    v8h ha;
    #pragma unroll
    for (int j = 0; j < 8; ++j) ha[j] = (_Float16)af[j];
    *(v8h*)(&As[ar * LSTR + akg]) = ha;          // one b128 store
    #pragma unroll
    for (int j = 0; j < 8; ++j) Bs[(bng + j) * LSTR + bkr] = bh[j];  // transpose
}

__device__ __forceinline__ void gemm_compute(
    const _Float16* __restrict__ As, const _Float16* __restrict__ Bs,
    int lane, int wm, int wn, v8f& acc0, v8f& acc1)
{
    // A fragment (16x32): lanes 0-15: M=lane, K 0-7 & 16-23;
    //                     lanes 16-31: M=lane-16, K 8-15 & 24-31
    const int am  = lane & 15;
    const int akb = (lane >> 4) << 3;            // 0 or 8
    const _Float16* Ar = &As[(wm * 16 + am) * LSTR + akb];
    const v8h alo = *(const v8h*)(Ar);
    const v8h ahi = *(const v8h*)(Ar + 16);
    v16h a;
    #pragma unroll
    for (int h = 0; h < 8; ++h) { a[h] = alo[h]; a[8 + h] = ahi[h]; }

    // B fragments (32x16): lane = N column, lanes 0-15 K 0-7 & 16-23, 16-31 K 8-15 & 24-31
    const int bn = lane & 15;
    const int bk = (lane >> 4) << 3;             // 0 or 8
    const _Float16* Br0 = &Bs[(wn * 32 + bn) * LSTR + bk];
    const _Float16* Br1 = &Bs[(wn * 32 + 16 + bn) * LSTR + bk];
    const v8h b0lo = *(const v8h*)(Br0);
    const v8h b0hi = *(const v8h*)(Br0 + 16);
    const v8h b1lo = *(const v8h*)(Br1);
    const v8h b1hi = *(const v8h*)(Br1 + 16);
    v16h bb0, bb1;
    #pragma unroll
    for (int h = 0; h < 8; ++h) {
        bb0[h] = b0lo[h]; bb0[8 + h] = b0hi[h];
        bb1[h] = b1lo[h]; bb1[8 + h] = b1hi[h];
    }

    acc0 = __builtin_amdgcn_wmma_f32_16x16x32_f16(
               false, a, false, bb0, (short)0, acc0, false, false);
    acc1 = __builtin_amdgcn_wmma_f32_16x16x32_f16(
               false, a, false, bb1, (short)0, acc1, false, false);
}

__global__ __launch_bounds__(256) void gemm_wmma_kernel(
    const float* __restrict__ W, const _Float16* __restrict__ Xf,
    const float* __restrict__ bias, float* __restrict__ Z,
    int O, int C, int N)
{
    __shared__ __align__(16) _Float16 As[2][64 * LSTR];   // [o(64)][k(32)+pad]
    __shared__ __align__(16) _Float16 Bs[2][64 * LSTR];   // [n(64)][k(32)+pad]

    const int tid  = threadIdx.x;
    const int b    = blockIdx.z;
    const int o0   = blockIdx.y * 64;
    const int n0   = blockIdx.x * 64;
    const int lane = tid & 31;
    const int wave = tid >> 5;
    const int wm   = wave >> 1;        // 0..3
    const int wn   = wave & 1;         // 0..1

    // staging coordinates
    const int ar  = tid >> 2;          // A: o row 0..63
    const int akg = (tid & 3) * 8;     // A: k group {0,8,16,24}
    const int bkr = tid >> 3;          // B: k row 0..31
    const int bng = (tid & 7) * 8;     // B: n group {0..56}

    v8f acc0 = {};
    v8f acc1 = {};

    const _Float16* Xb = Xf + (size_t)b * C * N;
    const float*    Wr = W + (size_t)(o0 + ar) * C;

    const int nt = (C + 31) >> 5;      // number of 32-wide K tiles

    float af[8];
    v8h   bh;
    gemm_load_regs(Wr, Xb, C, N, n0, 0, akg, bkr, bng, af, bh);
    gemm_stage_store(As[0], Bs[0], ar, akg, bkr, bng, af, bh);
    __syncthreads();

    int buf = 0;
    for (int t = 0; t < nt; ++t) {
        const bool more = (t + 1 < nt);
        if (more)
            gemm_load_regs(Wr, Xb, C, N, n0, (t + 1) * 32, akg, bkr, bng, af, bh);
        gemm_compute(As[buf], Bs[buf], lane, wm, wn, acc0, acc1);
        if (more)
            gemm_stage_store(As[buf ^ 1], Bs[buf ^ 1], ar, akg, bkr, bng, af, bh);
        __syncthreads();
        buf ^= 1;
    }

    // ---- epilogue: C/D layout VGPR r -> M = r + (lane>=16 ? 8 : 0), N = lane&15
    const int rm = (lane >> 4) << 3;
    const int cn = lane & 15;
    #pragma unroll
    for (int r = 0; r < 8; ++r) {
        const int o = o0 + wm * 16 + rm + r;
        const float bsv = bias[o];
        const size_t base = ((size_t)b * O + o) * N + n0 + wn * 32 + cn;
        Z[base]      = acc0[r] + bsv;
        Z[base + 16] = acc1[r] + bsv;
    }
}

// ===========================================================================
// BatchNorm over (B, N) per channel: one block per channel (Z fp32).
// ===========================================================================
__global__ __launch_bounds__(256) void bn_stats_kernel(
    const float* __restrict__ Z, int C, int N,
    float* __restrict__ mean, float* __restrict__ rstd)
{
    const int c = blockIdx.x;
    float s = 0.f, s2 = 0.f;
    for (int i = threadIdx.x; i < BB * N; i += 256) {
        const int b = i / N, n = i % N;
        const float v = Z[((size_t)b * C + c) * N + n];
        s += v; s2 += v * v;
    }
    __shared__ float S1[256], S2[256];
    S1[threadIdx.x] = s; S2[threadIdx.x] = s2;
    __syncthreads();
    for (int st = 128; st > 0; st >>= 1) {
        if ((int)threadIdx.x < st) {
            S1[threadIdx.x] += S1[threadIdx.x + st];
            S2[threadIdx.x] += S2[threadIdx.x + st];
        }
        __syncthreads();
    }
    if (threadIdx.x == 0) {
        const float inv = 1.f / (float)(BB * N);
        const float m = S1[0] * inv;
        const float var = S2[0] * inv - m * m;
        mean[c] = m;
        rstd[c] = rsqrtf(var + EPS);
    }
}

// BN + ReLU: read fp32 pre-activation, write f16 activation.
__global__ __launch_bounds__(256) void bn_relu_kernel(
    _Float16* __restrict__ A, const float* __restrict__ Z,
    const float* __restrict__ mean, const float* __restrict__ rstd,
    const float* __restrict__ g, const float* __restrict__ be, int C, int N)
{
    const long i = (long)blockIdx.x * 256 + threadIdx.x;
    if (i >= (long)BB * C * N) return;
    const int c = (int)((i / N) % C);
    float v = Z[i];
    v = g[c] * (v - mean[c]) * rstd[c] + be[c];
    A[i] = (_Float16)fmaxf(v, 0.f);
}

// pool[b,c] = mean_{s<64} afford[b, s, c]   (afford is fp32 (B, 576, 512))
__global__ __launch_bounds__(256) void pool_kernel(
    const float* __restrict__ aff, float* __restrict__ pool)
{
    const int i = blockIdx.x * 256 + threadIdx.x;   // b*512 + c
    if (i >= BB * EMB) return;
    const int b = i >> 9, c = i & 511;
    const float* p = aff + (size_t)b * 576 * EMB + c;
    float s = 0.f;
    for (int sidx = 0; sidx < NP; ++sidx) s += p[(size_t)sidx * EMB];
    pool[i] = s * (1.f / (float)NP);
}

__global__ __launch_bounds__(256) void scale_kernel(
    _Float16* __restrict__ U, const float* __restrict__ pool, int C, int N)
{
    const long i = (long)blockIdx.x * 256 + threadIdx.x;
    if (i >= (long)BB * C * N) return;
    const int c = (int)((i / N) % C);
    const int b = (int)(i / ((long)C * N));
    U[i] = (_Float16)((float)U[i] * pool[b * C + c]);
}

// ===========================================================================
// Head finalize: per point n, BN over (B, 64), ReLU, project w2, sigmoid.
// Zh fp32 (B, 64, N) with bias b1 already folded in by the GEMM.
// ===========================================================================
__global__ __launch_bounds__(256) void head_finalize_kernel(
    const float* __restrict__ Zh, const float* __restrict__ g,
    const float* __restrict__ be, const float* __restrict__ w2,
    const float* __restrict__ b2, float* __restrict__ out, int N)
{
    const int n = blockIdx.x;
    const int t = threadIdx.x;
    const int b = t >> 6;
    const int o = t & 63;
    const float v = Zh[((size_t)b * 64 + o) * N + n];
    __shared__ float S1[256], S2[256];
    S1[t] = v; S2[t] = v * v;
    __syncthreads();
    for (int st = 128; st > 0; st >>= 1) {
        if (t < st) { S1[t] += S1[t + st]; S2[t] += S2[t + st]; }
        __syncthreads();
    }
    const float m   = S1[0] * (1.f / 256.f);
    const float var = S2[0] * (1.f / 256.f) - m * m;
    const float ri  = rsqrtf(var + EPS);
    __syncthreads();
    float hn = g[n] * (v - m) * ri + be[n];
    hn = fmaxf(hn, 0.f);
    S1[t] = hn * w2[o];
    __syncthreads();
    for (int st = 32; st > 0; st >>= 1) {
        if (o < st) S1[t] += S1[t + st];
        __syncthreads();
    }
    if (o == 0) {
        const float z = S1[t] + b2[0];
        out[(size_t)b * N + n] = 1.f / (1.f + expf(-z));
    }
}

// ===========================================================================
// Host launcher
// ===========================================================================
static inline int nblk(long total) { return (int)((total + 255) / 256); }

extern "C" void kernel_launch(void* const* d_in, const int* in_sizes, int n_in,
                              void* d_out, int out_size, void* d_ws, size_t ws_size,
                              hipStream_t stream)
{
    const float* multi   = (const float*)d_in[0];
    const float* afford  = (const float*)d_in[1];
    const float* xyz0    = (const float*)d_in[2];
    const float* pts0    = (const float*)d_in[3];
    const float* xyz1    = (const float*)d_in[4];
    const float* pts1    = (const float*)d_in[5];
    const float* xyz2    = (const float*)d_in[6];
    const float* pts2    = (const float*)d_in[7];
    const float* xyz3    = (const float*)d_in[8];
    const float* fp3_w1  = (const float*)d_in[9];
    const float* fp3_b1  = (const float*)d_in[10];
    const float* fp3_g1  = (const float*)d_in[11];
    const float* fp3_be1 = (const float*)d_in[12];
    const float* fp3_w2  = (const float*)d_in[13];
    const float* fp3_b2  = (const float*)d_in[14];
    const float* fp3_g2  = (const float*)d_in[15];
    const float* fp3_be2 = (const float*)d_in[16];
    const float* fp2_w1  = (const float*)d_in[17];
    const float* fp2_b1  = (const float*)d_in[18];
    const float* fp2_g1  = (const float*)d_in[19];
    const float* fp2_be1 = (const float*)d_in[20];
    const float* fp2_w2  = (const float*)d_in[21];
    const float* fp2_b2  = (const float*)d_in[22];
    const float* fp2_g2  = (const float*)d_in[23];
    const float* fp2_be2 = (const float*)d_in[24];
    const float* fp1_w1  = (const float*)d_in[25];
    const float* fp1_b1  = (const float*)d_in[26];
    const float* fp1_g1  = (const float*)d_in[27];
    const float* fp1_be1 = (const float*)d_in[28];
    const float* fp1_w2  = (const float*)d_in[29];
    const float* fp1_b2  = (const float*)d_in[30];
    const float* fp1_g2  = (const float*)d_in[31];
    const float* fp1_be2 = (const float*)d_in[32];
    const float* head_w1 = (const float*)d_in[33];
    const float* head_b1 = (const float*)d_in[34];
    const float* head_g  = (const float*)d_in[35];
    const float* head_be = (const float*)d_in[36];
    const float* head_w2 = (const float*)d_in[37];
    const float* head_b2 = (const float*)d_in[38];
    float* out = (float*)d_out;

    // ---- workspace carving (256B aligned regions) ----
    char* w8 = (char*)d_ws;
    size_t p = 0;
    auto carve = [&](size_t bytes) -> char* {
        char* r = w8 + p;
        p += (bytes + 255) & ~(size_t)255;
        return r;
    };
    _Float16* X16 = (_Float16*)carve((size_t)BB * 518 * N0 * sizeof(_Float16)); // GEMM inputs
    float*    Z   = (float*)   carve((size_t)BB * 512 * N0 * sizeof(float));    // GEMM outputs
    _Float16* U16 = (_Float16*)carve((size_t)BB * 512 * N0 * sizeof(_Float16)); // stage outputs
    _Float16* P16 = (_Float16*)carve((size_t)BB * NP * EMB * sizeof(_Float16)); // P_align f16
    int*      KI  = (int*)     carve((size_t)BB * N0 * 3 * sizeof(int));
    float*    KW  = (float*)   carve((size_t)BB * N0 * 3 * sizeof(float));
    float*    MEAN= (float*)   carve(768 * sizeof(float));
    float*    RSTD= (float*)   carve(768 * sizeof(float));
    float*    POOL= (float*)   carve((size_t)BB * EMB * sizeof(float));

    convert_palign_kernel<<<nblk(BB * NP * EMB), 256, 0, stream>>>(multi, P16);

    // =========================== stage 3 (N2=256) ===========================
    knn3_kernel<<<dim3(NN2 / 256, 1, BB), 256, 0, stream>>>(xyz2, xyz3, NN2, NP, KI, KW);
    copy_channels_kernel<<<nblk((long)BB * 512 * NN2), 256, 0, stream>>>(X16, pts2, 512, NN2, 1024, 0);
    gather_interp_kernel<<<nblk((long)BB * 512 * NN2), 256, 0, stream>>>(
        X16, P16, KI, KW, 512, NN2, 1024, 512, (long)NP * EMB, 1L, (long)EMB);
    gemm_wmma_kernel<<<dim3(NN2 / 64, 768 / 64, BB), 256, 0, stream>>>(fp3_w1, X16, fp3_b1, Z, 768, 1024, NN2);
    bn_stats_kernel<<<768, 256, 0, stream>>>(Z, 768, NN2, MEAN, RSTD);
    bn_relu_kernel<<<nblk((long)BB * 768 * NN2), 256, 0, stream>>>(X16, Z, MEAN, RSTD, fp3_g1, fp3_be1, 768, NN2);
    gemm_wmma_kernel<<<dim3(NN2 / 64, 512 / 64, BB), 256, 0, stream>>>(fp3_w2, X16, fp3_b2, Z, 512, 768, NN2);
    bn_stats_kernel<<<512, 256, 0, stream>>>(Z, 512, NN2, MEAN, RSTD);
    bn_relu_kernel<<<nblk((long)BB * 512 * NN2), 256, 0, stream>>>(U16, Z, MEAN, RSTD, fp3_g2, fp3_be2, 512, NN2);

    // =========================== stage 2 (N1=1024) ==========================
    knn3_kernel<<<dim3(NN1 / 256, 1, BB), 256, 0, stream>>>(xyz1, xyz2, NN1, NN2, KI, KW);
    copy_channels_kernel<<<nblk((long)BB * 320 * NN1), 256, 0, stream>>>(X16, pts1, 320, NN1, 832, 0);
    gather_interp_kernel<<<nblk((long)BB * 512 * NN1), 256, 0, stream>>>(
        X16, U16, KI, KW, 512, NN1, 832, 320, (long)512 * NN2, (long)NN2, 1L);
    gemm_wmma_kernel<<<dim3(NN1 / 64, 768 / 64, BB), 256, 0, stream>>>(fp2_w1, X16, fp2_b1, Z, 768, 832, NN1);
    bn_stats_kernel<<<768, 256, 0, stream>>>(Z, 768, NN1, MEAN, RSTD);
    bn_relu_kernel<<<nblk((long)BB * 768 * NN1), 256, 0, stream>>>(X16, Z, MEAN, RSTD, fp2_g1, fp2_be1, 768, NN1);
    gemm_wmma_kernel<<<dim3(NN1 / 64, 512 / 64, BB), 256, 0, stream>>>(fp2_w2, X16, fp2_b2, Z, 512, 768, NN1);
    bn_stats_kernel<<<512, 256, 0, stream>>>(Z, 512, NN1, MEAN, RSTD);
    bn_relu_kernel<<<nblk((long)BB * 512 * NN1), 256, 0, stream>>>(U16, Z, MEAN, RSTD, fp2_g2, fp2_be2, 512, NN1);

    // =========================== stage 1 (N0=8192) ==========================
    knn3_kernel<<<dim3(N0 / 256, 1, BB), 256, 0, stream>>>(xyz0, xyz1, N0, NN1, KI, KW);
    copy_channels_kernel<<<nblk((long)BB * 3 * N0), 256, 0, stream>>>(X16, xyz0, 3, N0, 518, 0);
    copy_channels_kernel<<<nblk((long)BB * 3 * N0), 256, 0, stream>>>(X16, pts0, 3, N0, 518, 3);
    gather_interp_kernel<<<nblk((long)BB * 512 * N0), 256, 0, stream>>>(
        X16, U16, KI, KW, 512, N0, 518, 6, (long)512 * NN1, (long)NN1, 1L);
    gemm_wmma_kernel<<<dim3(N0 / 64, 512 / 64, BB), 256, 0, stream>>>(fp1_w1, X16, fp1_b1, Z, 512, 518, N0);
    bn_stats_kernel<<<512, 256, 0, stream>>>(Z, 512, N0, MEAN, RSTD);
    bn_relu_kernel<<<nblk((long)BB * 512 * N0), 256, 0, stream>>>(X16, Z, MEAN, RSTD, fp1_g1, fp1_be1, 512, N0);
    gemm_wmma_kernel<<<dim3(N0 / 64, 512 / 64, BB), 256, 0, stream>>>(fp1_w2, X16, fp1_b2, Z, 512, 512, N0);
    bn_stats_kernel<<<512, 256, 0, stream>>>(Z, 512, N0, MEAN, RSTD);
    bn_relu_kernel<<<nblk((long)BB * 512 * N0), 256, 0, stream>>>(U16, Z, MEAN, RSTD, fp1_g2, fp1_be2, 512, N0);

    // ============================== head ===================================
    pool_kernel<<<nblk(BB * EMB), 256, 0, stream>>>(afford, POOL);
    scale_kernel<<<nblk((long)BB * 512 * N0), 256, 0, stream>>>(U16, POOL, 512, N0);
    gemm_wmma_kernel<<<dim3(N0 / 64, 1, BB), 256, 0, stream>>>(head_w1, U16, head_b1, Z, 64, 512, N0);
    head_finalize_kernel<<<N0, 256, 0, stream>>>(Z, head_g, head_be, head_w2, head_b2, out, N0);
}
